// MHA_56427280335285
// MI455X (gfx1250) — matrix operand order repaired and checked by reading, not compiled, not run on hardware
//
#include <hip/hip_runtime.h>
#include <hip/hip_bf16.h>

#define D_MODEL 1024
#define SEQ     2048
#define BATCH   4
#define NHEAD   16
#define DK      64
#define ROWS    (BATCH * SEQ) /* 8192 */

typedef __attribute__((ext_vector_type(16))) __bf16 v16bf;
typedef __attribute__((ext_vector_type(8)))  __bf16 v8bf;
typedef __attribute__((ext_vector_type(8)))  float  v8f;

static __device__ __forceinline__ unsigned short f2bfbits(float f) {
  unsigned int u = __builtin_bit_cast(unsigned int, f);
  u += 0x7FFFu + ((u >> 16) & 1u);            // round-to-nearest-even
  return (unsigned short)(u >> 16);
}
static __device__ __forceinline__ __bf16 us2bf(unsigned short h) {
  return __builtin_bit_cast(__bf16, h);
}
// Assemble a 16-element bf16 fragment from two 16-byte chunks (b128 loads).
static __device__ __forceinline__ v16bf load_frag16(const unsigned short* p0,
                                                    const unsigned short* p1) {
  v8bf lo = *(const v8bf*)p0;
  v8bf hi = *(const v8bf*)p1;
  return __builtin_shufflevector(lo, hi, 0, 1, 2, 3, 4, 5, 6, 7,
                                         8, 9, 10, 11, 12, 13, 14, 15);
}

// ---------------------------------------------------------------------------
// Kernel 0: bulk fp32 -> bf16 conversion (bandwidth-bound pre-pass).
// 8 elements per thread: 2x float4 in, 1x uint4 (8 bf16) out.
// ---------------------------------------------------------------------------
__global__ __launch_bounds__(256) void cvt_bf16_kernel(
    const float* __restrict__ in, unsigned int* __restrict__ out, int n8)
{
  const int idx = blockIdx.x * blockDim.x + threadIdx.x;
  if (idx >= n8) return;
  const float4* p = (const float4*)in;
  const float4 a = p[idx * 2 + 0];
  const float4 b = p[idx * 2 + 1];
  uint4 o;
  o.x = (unsigned)f2bfbits(a.x) | ((unsigned)f2bfbits(a.y) << 16);
  o.y = (unsigned)f2bfbits(a.z) | ((unsigned)f2bfbits(a.w) << 16);
  o.z = (unsigned)f2bfbits(b.x) | ((unsigned)f2bfbits(b.y) << 16);
  o.w = (unsigned)f2bfbits(b.z) | ((unsigned)f2bfbits(b.w) << 16);
  ((uint4*)out)[idx] = o;
}

// ---------------------------------------------------------------------------
// Kernel 1: fused Q/K/V projection (xb @ Wb^T), all-bf16 WMMA + RoPE on Q,K.
// One wave computes a 16x64 output tile. blockIdx.z selects {Wq,Wk,Wv}.
// Q,K stored [B][H][S][DK]; V stored transposed [B][H][DK][S] so the flash
// kernel's P*V B-fragments are contiguous.
// ---------------------------------------------------------------------------
__global__ __launch_bounds__(128) void qkv_rope_kernel(
    const unsigned short* __restrict__ xb,
    const unsigned short* __restrict__ wqb,
    const unsigned short* __restrict__ wkb,
    const unsigned short* __restrict__ wvb,
    const int* __restrict__ tpos,
    unsigned short* __restrict__ qb, unsigned short* __restrict__ kb,
    unsigned short* __restrict__ vb)
{
  const int lane = threadIdx.x & 31;
  const int wave = threadIdx.x >> 5;
  const int m  = lane & 15;
  const int hi = lane >> 4;
  const int row0 = (blockIdx.x * 4 + wave) * 16;     // rows of x (B*S flat)
  const int col0 = blockIdx.y * 64;                  // output feature cols
  const int which = blockIdx.z;
  const unsigned short* w = (which == 0) ? wqb : (which == 1) ? wkb : wvb;

  v8f acc[4];
  #pragma unroll
  for (int t = 0; t < 4; ++t)
    #pragma unroll
    for (int j = 0; j < 8; ++j) acc[t][j] = 0.0f;

  for (int k0 = 0; k0 < D_MODEL; k0 += 32) {
    // A fragment: row = lane%16; halves 0-7 at k=hi*8, halves 8-15 at k=16+hi*8
    const unsigned short* ap = xb + (size_t)(row0 + m) * D_MODEL + k0;
    const v16bf a = load_frag16(ap + hi * 8, ap + 16 + hi * 8);
    #pragma unroll
    for (int t = 0; t < 4; ++t) {
      // B fragment: B[k][n] = W[n][k], halves 0-7 at k=hi*8, halves 8-15 at k=16+hi*8
      const int n = col0 + t * 16 + m;
      const unsigned short* bp = w + (size_t)n * D_MODEL + k0 + hi * 8;
      const v16bf b = load_frag16(bp, bp + 16);
      acc[t] = __builtin_amdgcn_wmma_f32_16x16x32_bf16(
          false, a, false, b, (short)0, acc[t], false, false);
    }
  }

  // RoPE (Q,K only) on f32 accumulator, then store bf16.
  #pragma unroll
  for (int t = 0; t < 4; ++t) {
    const int col = col0 + t * 16 + m;                 // lane's column
    const int h   = col >> 6;
    const int hd  = col & 63;
    const float ifreq = __expf(-(float)(hd & ~1) * (9.210340371976184f / 64.0f));
    #pragma unroll
    for (int j = 0; j < 8; ++j) {
      const int row  = row0 + 8 * hi + j;
      const int bidx = row >> 11;
      const int s    = row & (SEQ - 1);
      float c = acc[t][j];
      if (which == 2) {
        // V: store transposed [B][H][DK][S]
        vb[(((size_t)(bidx * NHEAD + h)) * DK + hd) * SEQ + s] = f2bfbits(c);
      } else {
        const float ang = (float)tpos[s] * ifreq;
        const float sn = __sinf(ang), cs = __cosf(ang);
        const float other = __shfl_xor(c, 1, 32);      // pair lives in lane^1
        const float r = ((hd & 1) == 0) ? (c * cs - other * sn)
                                        : (other * sn + c * cs);
        unsigned short* out = (which == 0) ? qb : kb;
        out[(((size_t)(bidx * NHEAD + h)) * SEQ + s) * DK + hd] = f2bfbits(r);
      }
    }
  }
}

// ---------------------------------------------------------------------------
// Kernel 2: causal flash attention, bf16 WMMA. One wave per 16 query rows of
// one (b,h). Key tiles of 32; online softmax; P re-laid-out via LDS.
// ---------------------------------------------------------------------------
__global__ __launch_bounds__(32) void flash_attn_kernel(
    const unsigned short* __restrict__ qb, const unsigned short* __restrict__ kb,
    const unsigned short* __restrict__ vb, unsigned short* __restrict__ ab)
{
  __shared__ __align__(16) unsigned short plds[16][32];  // 16x32 P tile
  const int lane = threadIdx.x & 31;
  const int m  = lane & 15;
  const int hi = lane >> 4;
  const int i0 = blockIdx.x * 16;                      // query row offset
  const size_t bh = (size_t)(blockIdx.z * NHEAD + blockIdx.y);
  const unsigned short* Q = qb + bh * SEQ * DK;
  const unsigned short* K = kb + bh * SEQ * DK;
  const unsigned short* V = vb + bh * DK * SEQ;        // transposed [DK][S]

  // Q fragments for k-dim 0..31 and 32..63 (kept resident).
  const unsigned short* qrow = Q + (size_t)(i0 + m) * DK;
  v16bf qf[2];
  #pragma unroll
  for (int ks = 0; ks < 2; ++ks)
    qf[ks] = load_frag16(qrow + ks * 32 + hi * 8, qrow + ks * 32 + 16 + hi * 8);

  float m_r[8], l_r[8];
  v8f o[4];
  #pragma unroll
  for (int j = 0; j < 8; ++j) { m_r[j] = -1e30f; l_r[j] = 0.0f; }
  #pragma unroll
  for (int t = 0; t < 4; ++t)
    #pragma unroll
    for (int j = 0; j < 8; ++j) o[t][j] = 0.0f;

  const int ntiles = (i0 + 16 + 31) / 32;
  for (int jt = 0; jt < ntiles; ++jt) {
    const int j0 = jt * 32;
    // logits tile 16x32 = 2 N-subtiles x 2 K-steps of WMMA
    v8f lg[2];
    #pragma unroll
    for (int ns = 0; ns < 2; ++ns) {
      #pragma unroll
      for (int j = 0; j < 8; ++j) lg[ns][j] = 0.0f;
      #pragma unroll
      for (int ks = 0; ks < 2; ++ks) {
        const unsigned short* kp =
            K + (size_t)(j0 + ns * 16 + m) * DK + ks * 32 + hi * 8;
        const v16bf kf = load_frag16(kp, kp + 16);
        lg[ns] = __builtin_amdgcn_wmma_f32_16x16x32_bf16(
            false, qf[ks], false, kf, (short)0, lg[ns], false, false);
      }
    }
    // ensure previous-iteration LDS reads are done before overwriting P
    asm volatile("s_wait_dscnt 0" ::: "memory");
    #pragma unroll
    for (int j = 0; j < 8; ++j) {
      const int qrowi = i0 + 8 * hi + j;
      float l0 = lg[0][j] * 0.125f;                 // 1/sqrt(64)
      float l1 = lg[1][j] * 0.125f;
      if (j0 + m > qrowi)      l0 = -1e30f;         // causal mask
      if (j0 + 16 + m > qrowi) l1 = -1e30f;
      float tmax = fmaxf(l0, l1);
      #pragma unroll
      for (int d = 8; d >= 1; d >>= 1) tmax = fmaxf(tmax, __shfl_xor(tmax, d, 32));
      const float mnew = fmaxf(m_r[j], tmax);
      const float p0 = __expf(l0 - mnew);
      const float p1 = __expf(l1 - mnew);
      float rs = p0 + p1;
      #pragma unroll
      for (int d = 8; d >= 1; d >>= 1) rs += __shfl_xor(rs, d, 32);
      const float corr = __expf(m_r[j] - mnew);
      l_r[j] = l_r[j] * corr + rs;
      m_r[j] = mnew;
      #pragma unroll
      for (int t = 0; t < 4; ++t) o[t][j] *= corr;  // rescale running output
      plds[8 * hi + j][m]      = f2bfbits(p0);      // C-layout -> LDS
      plds[8 * hi + j][16 + m] = f2bfbits(p1);
    }
    asm volatile("s_wait_dscnt 0" ::: "memory");
    // reload P as an A fragment (16x32 bf16)
    const v16bf pf = load_frag16(&plds[m][hi * 8], &plds[m][16 + hi * 8]);
    // O += P (16x32) x V (32x64) : 4 WMMAs; V^T makes B-fragments contiguous
    #pragma unroll
    for (int t = 0; t < 4; ++t) {
      const unsigned short* vp = V + (size_t)(t * 16 + m) * SEQ + j0 + hi * 8;
      const v16bf vf = load_frag16(vp, vp + 16);
      o[t] = __builtin_amdgcn_wmma_f32_16x16x32_bf16(
          false, pf, false, vf, (short)0, o[t], false, false);
    }
  }

  // normalize and store bf16 attn output [B*S][D_MODEL]
  #pragma unroll
  for (int j = 0; j < 8; ++j) {
    const float inv = 1.0f / l_r[j];
    const size_t row = (size_t)blockIdx.z * SEQ + i0 + 8 * hi + j;
    #pragma unroll
    for (int t = 0; t < 4; ++t)
      ab[row * D_MODEL + (size_t)blockIdx.y * DK + t * 16 + m] =
          f2bfbits(o[t][j] * inv);
  }
}

// ---------------------------------------------------------------------------
// Kernel 3: output projection attn @ Wo^T -> fp32 d_out. Same tiling as k1.
// ---------------------------------------------------------------------------
__global__ __launch_bounds__(128) void oproj_kernel(
    const unsigned short* __restrict__ ab, const unsigned short* __restrict__ wob,
    float* __restrict__ out)
{
  const int lane = threadIdx.x & 31;
  const int wave = threadIdx.x >> 5;
  const int m  = lane & 15;
  const int hi = lane >> 4;
  const int row0 = (blockIdx.x * 4 + wave) * 16;
  const int col0 = blockIdx.y * 64;

  v8f acc[4];
  #pragma unroll
  for (int t = 0; t < 4; ++t)
    #pragma unroll
    for (int j = 0; j < 8; ++j) acc[t][j] = 0.0f;

  for (int k0 = 0; k0 < D_MODEL; k0 += 32) {
    const unsigned short* ap = ab + (size_t)(row0 + m) * D_MODEL + k0;
    const v16bf a = load_frag16(ap + hi * 8, ap + 16 + hi * 8);
    #pragma unroll
    for (int t = 0; t < 4; ++t) {
      const int n = col0 + t * 16 + m;
      const unsigned short* bp = wob + (size_t)n * D_MODEL + k0 + hi * 8;
      const v16bf b = load_frag16(bp, bp + 16);
      acc[t] = __builtin_amdgcn_wmma_f32_16x16x32_bf16(
          false, a, false, b, (short)0, acc[t], false, false);
    }
  }
  #pragma unroll
  for (int t = 0; t < 4; ++t)
    #pragma unroll
    for (int j = 0; j < 8; ++j)
      out[(size_t)(row0 + 8 * hi + j) * D_MODEL + col0 + t * 16 + m] = acc[t][j];
}

extern "C" void kernel_launch(void* const* d_in, const int* in_sizes, int n_in,
                              void* d_out, int out_size, void* d_ws, size_t ws_size,
                              hipStream_t stream) {
  const float* x  = (const float*)d_in[0];
  const float* wq = (const float*)d_in[1];
  const float* wk = (const float*)d_in[2];
  const float* wv = (const float*)d_in[3];
  const float* wo = (const float*)d_in[4];
  const int*   tp = (const int*)d_in[5];

  // workspace layout (bf16 elements):
  //   xb   : ROWS*D_MODEL                (16 MB)
  //   wqb/wkb/wvb/wob : D_MODEL*D_MODEL  ( 2 MB each)
  //   qb/kb/vb : ROWS*D_MODEL            (16 MB each)
  //   ab   : ROWS*D_MODEL                (16 MB)          total 88 MB
  unsigned short* xb  = (unsigned short*)d_ws;
  unsigned short* wqb = xb  + (size_t)ROWS * D_MODEL;
  unsigned short* wkb = wqb + (size_t)D_MODEL * D_MODEL;
  unsigned short* wvb = wkb + (size_t)D_MODEL * D_MODEL;
  unsigned short* wob = wvb + (size_t)D_MODEL * D_MODEL;
  unsigned short* qb  = wob + (size_t)D_MODEL * D_MODEL;
  unsigned short* kb  = qb  + (size_t)ROWS * D_MODEL;
  unsigned short* vb  = kb  + (size_t)ROWS * D_MODEL;
  unsigned short* ab  = vb  + (size_t)ROWS * D_MODEL;

  const int nx8 = ROWS * D_MODEL / 8;      // 1,048,576
  const int nw8 = D_MODEL * D_MODEL / 8;   // 131,072
  cvt_bf16_kernel<<<nx8 / 256, 256, 0, stream>>>(x,  (unsigned int*)xb,  nx8);
  cvt_bf16_kernel<<<nw8 / 256, 256, 0, stream>>>(wq, (unsigned int*)wqb, nw8);
  cvt_bf16_kernel<<<nw8 / 256, 256, 0, stream>>>(wk, (unsigned int*)wkb, nw8);
  cvt_bf16_kernel<<<nw8 / 256, 256, 0, stream>>>(wv, (unsigned int*)wvb, nw8);
  cvt_bf16_kernel<<<nw8 / 256, 256, 0, stream>>>(wo, (unsigned int*)wob, nw8);

  dim3 g1(ROWS / 64, D_MODEL / 64, 3);
  qkv_rope_kernel<<<g1, 128, 0, stream>>>(xb, wqb, wkb, wvb, tp, qb, kb, vb);

  dim3 g2(SEQ / 16, NHEAD, BATCH);
  flash_attn_kernel<<<g2, 32, 0, stream>>>(qb, kb, vb, ab);

  dim3 g3(ROWS / 64, D_MODEL / 64, 1);
  oproj_kernel<<<g3, 128, 0, stream>>>(ab, wob, (float*)d_out);
}
